// Encoder_73452530696378
// MI455X (gfx1250) — matrix operand order repaired and checked by reading, not compiled, not run on hardware
//
#include <hip/hip_runtime.h>
#include <hip/hip_bf16.h>

typedef __bf16 bf16_t;
typedef __attribute__((ext_vector_type(16))) __bf16 v16bf;
typedef __attribute__((ext_vector_type(8)))  float  v8f;

#define B_DIM 2048
#define T_DIM 128
#define N_DIM 128
#define H_DIM 256
#define G_DIM 1024   // 4*H
#define BM    32     // batch rows per workgroup

__device__ __forceinline__ float sigmoid_f(float x) { return 1.0f / (1.0f + __expf(-x)); }

// ---------------------------------------------------------------------------
// Prologue 1: pack f32 weights -> bf16 in WMMA B-operand tile layout.
// B tile (32x16, K x N) for v_wmma_f32_16x16x32_bf16: lane holds col = lane&15,
// K = e + 8*(lane>>4) + (e>=8 ? 8 : 0). Storage: [jt][kt][lane][e] contiguous -> one 32B read/lane.
//   wih_pk: 64 jt * 4 kt * 32 * 16 = 131072 bf16 (256 KB)
//   whh_pk: 64 jt * 8 kt * 32 * 16 = 262144 bf16 (512 KB)
// ---------------------------------------------------------------------------
__global__ __launch_bounds__(256) void pack_weights_kernel(const float* __restrict__ w_ih,
                                                           const float* __restrict__ w_hh,
                                                           bf16_t* __restrict__ wih_pk,
                                                           bf16_t* __restrict__ whh_pk) {
  const int idx = blockIdx.x * 256 + threadIdx.x;   // 0 .. 262143
  if (idx < 131072) {
    const int e = idx & 15, lane = (idx >> 4) & 31, kt = (idx >> 9) & 3, jt = idx >> 11;
    const int col = lane & 15, hi = lane >> 4;
    wih_pk[idx] = (bf16_t)w_ih[(jt * 16 + col) * N_DIM + kt * 32 + hi * 8 + e + ((e >= 8) ? 8 : 0)];
  }
  {
    const int e = idx & 15, lane = (idx >> 4) & 31, kt = (idx >> 9) & 7, jt = idx >> 12;
    const int col = lane & 15, hi = lane >> 4;
    whh_pk[idx] = (bf16_t)w_hh[(jt * 16 + col) * H_DIM + kt * 32 + hi * 8 + e + ((e >= 8) ? 8 : 0)];
  }
}

// ---------------------------------------------------------------------------
// Prologue 2: precomp[b,n] = sum_t input[b,t,n] * w_x[t]
// ---------------------------------------------------------------------------
__global__ __launch_bounds__(128) void precomp_kernel(const float* __restrict__ input,
                                                      const float* __restrict__ w_attn,
                                                      float* __restrict__ precomp) {
  __shared__ float wx_s[T_DIM];
  const int b = blockIdx.x;
  const int n = threadIdx.x;
  wx_s[n] = w_attn[2 * H_DIM + n];
  __syncthreads();
  float acc = 0.0f;
  const float* p = input + (long)b * T_DIM * N_DIM + n;
#pragma unroll 8
  for (int t = 0; t < T_DIM; ++t) acc += p[(long)t * N_DIM] * wx_s[t];
  precomp[(long)b * N_DIM + n] = acc;
}

// ---------------------------------------------------------------------------
// Main persistent recurrent kernel: 64 WGs x 256 threads, each WG owns 32
// batch rows across all T steps. State in LDS; GEMMs via bf16 WMMA.
// A-operand packed layout for a 16x32 bf16 tile: lane slot = hi*16 + row,
// element e maps to K = (v<4 ? 2v : 16+2(v-4)) + (e&1) + 8*hi, v=e>>1.
// ---------------------------------------------------------------------------
__global__ __launch_bounds__(256) void encoder_main_kernel(
    const float* __restrict__ input, const bf16_t* __restrict__ wih_pk,
    const bf16_t* __restrict__ whh_pk, const float* __restrict__ precomp,
    const float* __restrict__ w_attn, const float* __restrict__ b_attn,
    const float* __restrict__ b_ih, const float* __restrict__ b_hh,
    float* __restrict__ out_w, float* __restrict__ out_e) {
  extern __shared__ char smem[];
  float*  c_lds  = (float*)smem;            // 8192 f32  : c[32][256]
  float*  s_lds  = c_lds + 8192;            // 32 f32
  float*  wh_lds = s_lds + 32;              // 256 f32
  float*  wc_lds = wh_lds + 256;            // 256 f32
  float*  gb_lds = wc_lds + 256;            // 1024 f32  : b_ih + b_hh
  bf16_t* h_pk   = (bf16_t*)(gb_lds + 1024);// 2 x 8192 bf16 (double buffer, A-packed)
  bf16_t* wx_pk  = h_pk + 16384;            // 4096 bf16 (A-packed)

  const int tid  = threadIdx.x;
  const int wave = tid >> 5;
  const int lane = tid & 31;
  const int b0   = blockIdx.x * BM;

  for (int i = tid; i < 8192; i += 256) c_lds[i] = 0.0f;
  for (int i = tid; i < 16384; i += 256) h_pk[i] = (bf16_t)0.0f;
  if (tid < 32) s_lds[tid] = 0.0f;
  if (tid < 256) { wh_lds[tid] = w_attn[tid]; wc_lds[tid] = w_attn[H_DIM + tid]; }
  for (int i = tid; i < 1024; i += 256) gb_lds[i] = b_ih[i] + b_hh[i];
  const float attn_bias = b_attn[0];
  __syncthreads();

  int buf = 0;
#pragma unroll 1
  for (int t = 0; t < T_DIM; ++t) {
    // ---- Phase A: s[b] = h . w_h + c . w_c ----
    if (tid < 32) s_lds[tid] = 0.0f;
    __syncthreads();
    {
      const int row = tid >> 3;          // 0..31
      const int seg = tid & 7;           // K tile (32 wide)
      const int mt = row >> 4, r = row & 15;
      float partial = 0.0f;
      const bf16_t* hb = h_pk + buf * 8192 + ((mt * 8 + seg) * 32) * 16;
#pragma unroll
      for (int hi = 0; hi < 2; ++hi) {
#pragma unroll
        for (int e = 0; e < 16; ++e) {
          const int v = e >> 1;
          const int koff = ((v < 4) ? (2 * v) : (16 + 2 * (v - 4))) + (e & 1) + 8 * hi;
          partial += (float)hb[(hi * 16 + r) * 16 + e] * wh_lds[seg * 32 + koff];
        }
      }
      const float* cr = c_lds + row * H_DIM + seg * 32;
#pragma unroll
      for (int i = 0; i < 32; ++i) partial += cr[i] * wc_lds[seg * 32 + i];
      atomicAdd(&s_lds[row], partial);
    }
    __syncthreads();

    // ---- Phase B: softmax over N, wx = attn * x_t ----
    {
      const int row = (wave << 2) + (lane >> 3);   // 0..31, 8 lanes per row
      const int n0  = (lane & 7) * 16;             // 16 cols per lane
      const long gbase = ((long)(b0 + row) * T_DIM + t) * N_DIM;
      const float sv = s_lds[row] + attn_bias;
      const float* xp = input + gbase + n0;
      const float* pp = precomp + (long)(b0 + row) * N_DIM + n0;
      float xv[16], lg[16];
      float mx = -3.0e38f;
#pragma unroll
      for (int j = 0; j < 16; ++j) {
        xv[j] = xp[j];
        lg[j] = sv + pp[j];
        mx = fmaxf(mx, lg[j]);
      }
#pragma unroll
      for (int m = 4; m >= 1; m >>= 1) mx = fmaxf(mx, __shfl_xor(mx, m, 8));
      float sum = 0.0f;
#pragma unroll
      for (int j = 0; j < 16; ++j) { lg[j] = __expf(lg[j] - mx); sum += lg[j]; }
#pragma unroll
      for (int m = 4; m >= 1; m >>= 1) sum += __shfl_xor(sum, m, 8);
      const float inv = 1.0f / sum;
      const int mt = row >> 4, r = row & 15;
#pragma unroll
      for (int j = 0; j < 16; ++j) {
        const int n = n0 + j;
        const float w = lg[j] * inv * xv[j];
        out_w[gbase + n] = w;
        const int kt = n >> 5;
        const int hi = (n >> 3) & 1;
        const int v  = ((n & 7) >> 1) + (((n & 31) >= 16) ? 4 : 0);
        const int e  = 2 * v + (n & 1);
        wx_pk[((mt * 4 + kt) * 32 + hi * 16 + r) * 16 + e] = (bf16_t)w;
      }
      if (t + 1 < T_DIM) __builtin_prefetch(xp + N_DIM, 0, 0);
    }
    __syncthreads();

    // ---- Phase C: gates = wx @ w_ih^T + h @ w_hh^T + bias ; LSTM update ----
    {
      const int kbase = wave * 32;   // each wave owns 32 hidden cols
      const v16bf* wxv = (const v16bf*)wx_pk;
      const v16bf* hv  = (const v16bf*)(h_pk + buf * 8192);
#pragma unroll
      for (int ks = 0; ks < 2; ++ks) {
        const int k0 = kbase + ks * 16;
        v8f acc[2][4];
#pragma unroll
        for (int m = 0; m < 2; ++m)
#pragma unroll
          for (int g = 0; g < 4; ++g)
#pragma unroll
            for (int e = 0; e < 8; ++e) acc[m][g][e] = 0.0f;

#pragma unroll
        for (int g = 0; g < 4; ++g) {
          // Opaque zero: re-materialized every unrolled iteration so the
          // compiler cannot hoist/CSE the A/B fragment loads across g, ks or
          // the t loop (prevents the giant live range -> scratch spills).
          int az;
          asm volatile("s_mov_b32 %0, 0" : "=s"(az));
          const v16bf* bih = (const v16bf*)wih_pk + az;
          const v16bf* bhh = (const v16bf*)whh_pk + az;
          const v16bf* wxa = wxv + az;
          const v16bf* ha  = hv + az;
          const int jt = (g * 256 + k0) >> 4;
#pragma unroll
          for (int kt = 0; kt < 4; ++kt) {          // K over N_DIM=128 (wx)
            const v16bf bt = bih[(jt * 4 + kt) * 32 + lane];
#pragma unroll
            for (int m = 0; m < 2; ++m) {
              const v16bf at = wxa[(m * 4 + kt) * 32 + lane];
              acc[m][g] = __builtin_amdgcn_wmma_f32_16x16x32_bf16(
                  false, at, false, bt, (short)0, acc[m][g], false, false);
            }
          }
#pragma unroll
          for (int kt = 0; kt < 8; ++kt) {          // K over H_DIM=256 (h)
            const v16bf bt = bhh[(jt * 8 + kt) * 32 + lane];
#pragma unroll
            for (int m = 0; m < 2; ++m) {
              const v16bf at = ha[(m * 8 + kt) * 32 + lane];
              acc[m][g] = __builtin_amdgcn_wmma_f32_16x16x32_bf16(
                  false, at, false, bt, (short)0, acc[m][g], false, false);
            }
          }
          const float gb = gb_lds[g * 256 + k0 + (lane & 15)];
#pragma unroll
          for (int m = 0; m < 2; ++m)
#pragma unroll
            for (int e = 0; e < 8; ++e) acc[m][g][e] += gb;
        }

        // activations + state update (C/D layout: row = e + 8*(lane>=16), col = lane&15)
        const int col = k0 + (lane & 15);
        const int ebase = (lane >> 4) * 8;
#pragma unroll
        for (int m = 0; m < 2; ++m) {
#pragma unroll
          for (int e = 0; e < 8; ++e) {
            const int row = m * 16 + ebase + e;
            const float iv = acc[m][0][e];
            const float fv = acc[m][1][e];
            const float gv = acc[m][2][e];
            const float ov = acc[m][3][e];
            const float co = c_lds[row * H_DIM + col];
            const float cn = sigmoid_f(fv) * co + sigmoid_f(iv) * tanhf(gv);
            const float hn = sigmoid_f(ov) * tanhf(cn);
            c_lds[row * H_DIM + col] = cn;
            const int mt = row >> 4, r = row & 15;
            const int kt2 = col >> 5;
            const int hi = (col >> 3) & 1;
            const int v  = ((col & 7) >> 1) + (((col & 31) >= 16) ? 4 : 0);
            const int ee = 2 * v + (col & 1);
            h_pk[(buf ^ 1) * 8192 + ((mt * 8 + kt2) * 32 + hi * 16 + r) * 16 + ee] = (bf16_t)hn;
            out_e[((long)(b0 + row) * T_DIM + t) * H_DIM + col] = hn;
          }
        }
      }
    }
    buf ^= 1;
    __syncthreads();
  }
}

extern "C" void kernel_launch(void* const* d_in, const int* in_sizes, int n_in,
                              void* d_out, int out_size, void* d_ws, size_t ws_size,
                              hipStream_t stream) {
  (void)in_sizes; (void)n_in; (void)out_size; (void)ws_size;
  const float* input  = (const float*)d_in[0];
  const float* w_ih   = (const float*)d_in[1];
  const float* w_hh   = (const float*)d_in[2];
  const float* b_ih   = (const float*)d_in[3];
  const float* b_hh   = (const float*)d_in[4];
  const float* w_attn = (const float*)d_in[5];
  const float* b_attn = (const float*)d_in[6];

  float* out_w = (float*)d_out;                                   // (B,T,N)
  float* out_e = out_w + (size_t)B_DIM * T_DIM * N_DIM;           // (B,T,H)

  char* ws = (char*)d_ws;
  bf16_t* wih_pk  = (bf16_t*)ws;                                  // 256 KB
  bf16_t* whh_pk  = (bf16_t*)(ws + 262144);                       // 512 KB
  float*  precomp = (float*)(ws + 262144 + 524288);               // 1 MB

  pack_weights_kernel<<<1024, 256, 0, stream>>>(w_ih, w_hh, wih_pk, whh_pk);
  precomp_kernel<<<B_DIM, 128, 0, stream>>>(input, w_attn, precomp);

  const int smem_bytes = (8192 + 32 + 256 + 256 + 1024) * 4 + (16384 + 4096) * 2; // 80000
  encoder_main_kernel<<<B_DIM / BM, 256, smem_bytes, stream>>>(
      input, wih_pk, whh_pk, precomp, w_attn, b_attn, b_ih, b_hh, out_w, out_e);
}
